// MinConv2dGRU_72267119723115
// MI455X (gfx1250) — matrix operand order repaired and checked by reading, not compiled, not run on hardware
//
#include <hip/hip_runtime.h>
#include <hip/hip_bf16.h>

typedef __attribute__((ext_vector_type(16))) _Float16 v16h;
typedef __attribute__((ext_vector_type(8)))  _Float16 v8h;
typedef __attribute__((ext_vector_type(8)))  float    v8f;

// Geometry (fixed by the reference): B=4, S=32, H=W=32, hid=128, O=2*hid=256.
#define IMG   1024          // H*W
#define NIMG  128           // B*S
#define OCH   256
#define HID   128
#define BSTR  40            // LDS row stride (halves) for B tile (pixel-major)

// One-time weight re-layout + f16 convert:
//   w (O, Cin, 3, 3) f32  ->  wT (O, 9, Cin) f16   (GEMM K = rs*Cin + c)
__global__ __launch_bounds__(256)
void cvt_weights(const float* __restrict__ w, _Float16* __restrict__ wT,
                 int Cin, int total) {
    int idx = blockIdx.x * 256 + threadIdx.x;
    if (idx >= total) return;
    int o  = idx / (9 * Cin);
    int r2 = idx - o * (9 * Cin);
    int rs = r2 / Cin;
    int c  = r2 - rs * Cin;
    wT[idx] = (_Float16)w[(o * Cin + c) * 9 + rs];
}

// Implicit-GEMM 3x3 'same' conv, f16 WMMA, fp32 accumulate.
// K = rs*Cin + c: each 32-wide K chunk = 32 channels at one scalar (r,s).
// Block = 256 threads = 8 waves; block tile 128 outchans x 32 pixels (one row).
// Each wave owns a 16-row A fragment (streamed from global f16 wT, L2-resident)
// and computes both 16-col N subtiles: 2 WMMAs per chunk on one A operand.
// B tile is double-buffered in LDS: one barrier per K chunk.
__global__ __launch_bounds__(256)
void conv3x3_wmma(const float* __restrict__ in,      // (n, Cin, 32, 32) f32
                  const _Float16* __restrict__ wT,   // (256, 9, Cin) f16
                  const float* __restrict__ bias,    // (256,) f32
                  _Float16* __restrict__ gh,         // (n, 256, 32, 32) f16
                  int Cin, int cshift) {             // cshift = log2(Cin)
    const int Ktot  = Cin * 9;
    const int row   = blockIdx.x;           // output image row h
    const int oBase = blockIdx.y * 128;     // outchan tile base (0 or 128)
    const int n     = blockIdx.z;           // image index
    const int tid   = threadIdx.x;
    const int wave  = tid >> 5;             // 0..7 = 16-outchan M subtile
    const int lane  = tid & 31;
    const int hiH   = lane >> 4;            // lane half (0: lanes 0-15, 1: 16-31)
    const int l15   = lane & 15;
    const int p     = tid & 31;             // staging: pixel (invariant in chunk)
    const int kb    = tid >> 5;             // staging: base channel-in-chunk (0..7)

    __shared__ __align__(16) _Float16 Bsm[2][32 * BSTR];   // double-buffered [pixel][k]

    const float* __restrict__ inN = in + (size_t)n * Cin * IMG;
    // This lane's A-fragment row (outchan), K-contiguous in wT:
    const _Float16* __restrict__ wRow =
        wT + (size_t)(oBase + wave * 16 + l15) * Ktot;

    v8f acc0 = {};   // columns 0..15
    v8f acc1 = {};   // columns 16..31

#pragma unroll 2
    for (int K0 = 0; K0 < Ktot; K0 += 32) {
        _Float16* __restrict__ Bcur = Bsm[(K0 >> 5) & 1];

        // Scalar (r,s) for this chunk (uniform -> SALU).
        const int rs = K0 >> cshift;
        const int c0 = K0 & (Cin - 1);
        const int r  = rs / 3;
        const int s  = rs - r * 3;
        const int ih = row + r - 1;

        // ---- stage B: 32 channels x 32 pixels of one shifted input row ----
        // Predicate depends only on (p, s): one clamped base pointer, then the
        // 4 channel loads are immediate offsets (i*32KB) -> single load clause.
        {
            const int  iw = p + s - 1;
            const bool ok = ((unsigned)ih < 32u) && ((unsigned)iw < 32u);
            const float* rowPtr  = inN + (((size_t)c0) << 10) + (ih << 5) + iw;
            const float* basePtr = ok ? rowPtr : inN;   // inN+k*4KB always mapped
            float vals[4];
#pragma unroll
            for (int i = 0; i < 4; ++i) {
                vals[i] = basePtr[((size_t)(kb + i * 8)) << 10];
                // Speculative L2 prefetch of next K chunk (+32 channels);
                // dropped on translation failure, so tail overrun is harmless.
                __builtin_prefetch(rowPtr + (((size_t)(kb + i * 8 + 32)) << 10), 0, 1);
            }
#pragma unroll
            for (int i = 0; i < 4; ++i)
                Bcur[p * BSTR + kb + i * 8] = (_Float16)(ok ? vals[i] : 0.0f);
        }
        __syncthreads();   // staging visible; also guards WAR on the other buffer

        // ---- A fragment: two 16B global loads (contiguous k order, same as B) ----
        // lanes 0-15 (row M=l15) hold K0..15; lanes 16-31 hold K16..31
        const _Float16* aptr = wRow + K0 + hiH * 16;
        v8h a0 = *(const v8h*)(aptr);
        v8h a1 = *(const v8h*)(aptr + 8);
        // ---- B fragments from LDS: lanes 0-15 K0..15, lanes 16-31 K16..31 ----
        const _Float16* bcolL = &Bcur[l15 * BSTR + hiH * 16];          // cols 0..15
        const _Float16* bcolH = &Bcur[(16 + l15) * BSTR + hiH * 16];   // cols 16..31
        v8h bl0 = *(const v8h*)(bcolL);
        v8h bl1 = *(const v8h*)(bcolL + 8);
        v8h bh0 = *(const v8h*)(bcolH);
        v8h bh1 = *(const v8h*)(bcolH + 8);

        v16h a, bL, bH;
#pragma unroll
        for (int i = 0; i < 8; ++i) {
            a[i]  = a0[i];   a[i + 8]  = a1[i];
            bL[i] = bl0[i];  bL[i + 8] = bl1[i];
            bH[i] = bh0[i];  bH[i + 8] = bh1[i];
        }

        acc0 = __builtin_amdgcn_wmma_f32_16x16x32_f16(
            false, a, false, bL, (short)0, acc0, false, false);
        acc1 = __builtin_amdgcn_wmma_f32_16x16x32_f16(
            false, a, false, bH, (short)0, acc1, false, false);
        // no trailing barrier: next chunk stages the other LDS buffer
    }

    // ---- store D (+bias): VGPR v -> M = v + 8*hiH, N = l15 / 16+l15 ----
#pragma unroll
    for (int v = 0; v < 8; ++v) {
        int o = oBase + wave * 16 + hiH * 8 + v;
        float bo = bias[o];
        size_t base = ((size_t)n * OCH + o) * IMG + row * 32;
        gh[base + l15]      = (_Float16)(acc0[v] + bo);
        gh[base + 16 + l15] = (_Float16)(acc1[v] + bo);
    }
}

// Sequential minGRU scan over S=32: h_t = h + z*(g - h), z = sigmoid(gate),
// g(x) = x + 0.5 (x>=0) else sigmoid(x). h0 = 0.5. One thread per (b, c, hw).
__global__ __launch_bounds__(256)
void mingru_scan(const _Float16* __restrict__ gh,   // (B*S, 256, 1024) f16
                 float* __restrict__ out,           // (B*S, 128, 1024) f32
                 float* __restrict__ fin) {         // (B, 128, 1024)  f32
    int idx = blockIdx.x * 256 + threadIdx.x;       // 0 .. 524287
    int hw = idx & (IMG - 1);
    int c  = (idx >> 10) & (HID - 1);
    int b  = idx >> 17;

    float h = 0.5f;
#pragma unroll 4
    for (int t = 0; t < 32; ++t) {
        size_t base = (size_t)(b * 32 + t) * OCH * IMG;
        float gate = (float)gh[base + (size_t)c * IMG + hw];
        float hd   = (float)gh[base + (size_t)(c + HID) * IMG + hw];
        float z = 1.0f / (1.0f + __expf(-gate));
        float g = (hd >= 0.0f) ? (hd + 0.5f) : (1.0f / (1.0f + __expf(-hd)));
        h = h + z * (g - h);
        out[((size_t)(b * 32 + t) * HID + c) * IMG + hw] = h;
    }
    fin[((size_t)b * HID + c) * IMG + hw] = h;
}

extern "C" void kernel_launch(void* const* d_in, const int* in_sizes, int n_in,
                              void* d_out, int out_size, void* d_ws, size_t ws_size,
                              hipStream_t stream) {
    const float* x  = (const float*)d_in[0];   // (4,32,64,32,32)
    const float* w0 = (const float*)d_in[1];   // (256,64,3,3)
    const float* b0 = (const float*)d_in[2];   // (256,)
    const float* w1 = (const float*)d_in[3];   // (256,128,3,3)
    const float* b1 = (const float*)d_in[4];   // (256,)

    float* out  = (float*)d_out;               // (4,32,128,32,32) = 16,777,216 f32
    float* fin0 = out + 16777216;              // finals layer 0: 524,288 f32
    float* fin1 = fin0 + 524288;               // finals layer 1: 524,288 f32

    _Float16* gh  = (_Float16*)d_ws;           // (128,256,32,32) f16 = 64 MB
    _Float16* wT0 = gh + 33554432;             // 147,456 halves
    _Float16* wT1 = wT0 + 147456;              // 294,912 halves
    float* h1 = out;                           // layer-0 activations live in the (dead)
                                               // output region until the final scan.

    // Weight re-layout + f16 convert (tiny, one-time per call).
    cvt_weights<<<(147456 + 255) / 256, 256, 0, stream>>>(w0, wT0, 64, 147456);
    cvt_weights<<<(294912 + 255) / 256, 256, 0, stream>>>(w1, wT1, 128, 294912);

    dim3 cgrid(32, 2, NIMG), cblk(256);
    // Layer 0: conv (Cin=64) -> gh, scan -> h1 (+finals[0])
    conv3x3_wmma<<<cgrid, cblk, 0, stream>>>(x, wT0, b0, gh, 64, 6);
    mingru_scan<<<524288 / 256, 256, 0, stream>>>(gh, h1, fin0);
    // Layer 1: conv (Cin=128) over h1 -> gh, scan -> out (+finals[1])
    conv3x3_wmma<<<cgrid, cblk, 0, stream>>>(h1, wT1, b1, gh, 128, 7);
    mingru_scan<<<524288 / 256, 256, 0, stream>>>(gh, out, fin1);
}
